// ProbSmoothAttentionPool_36919538876521
// MI455X (gfx1250) — matrix operand order repaired and checked by reading, not compiled, not run on hardware
//
#include <hip/hip_runtime.h>
#include <hip/hip_bf16.h>
#include <math.h>

// ---------------- problem constants (from the reference) ----------------
#define B_DIM   8
#define N_DIM   2048
#define D_DIM   256
#define A2_DIM  256
#define S_DIM   1000
#define SCH     64
#define KPAD    72          // prob-tile row stride in halves (144 B, 16B-aligned)

typedef _Float16 v16h __attribute__((ext_vector_type(16)));
typedef _Float16 v8h  __attribute__((ext_vector_type(8)));
typedef float    v8f  __attribute__((ext_vector_type(8)));

__device__ __forceinline__ float gelu_exact(float x) {
    return 0.5f * x * (1.0f + erff(x * 0.70710678118654752440f));
}

__device__ __forceinline__ v16h cat8(v8h lo, v8h hi8) {
    return __builtin_shufflevector(lo, hi8, 0, 1, 2, 3, 4, 5, 6, 7,
                                            8, 9, 10, 11, 12, 13, 14, 15);
}

// CDNA5 16-bit A-matrix (16x32) per-lane K index (v = vgpr 0..7, h = half):
//   lanes 0-15: VGPR0..3 -> K 0..7, VGPR4..7 -> K 16..23 ; lanes16-31: +8
__device__ __forceinline__ int a_k(int v, int hi) {
    int kb = (v < 4) ? (2 * v) : (16 + 2 * (v - 4));
    return kb + (hi ? 8 : 0);
}

// =====================================================================
// Pack W (256x256 f32, row-major [k][n]) into WMMA B-operand layout:
//   tile t = j*8+ks ; lane's 16 halves are K = 8*hi + (0..7), 16 + 8*hi + (0..7), N = j*16+r
// =====================================================================
__global__ __launch_bounds__(256)
void pack_w_kernel(const float* __restrict__ W, _Float16* __restrict__ out)
{
    const int gid  = blockIdx.x * 256 + threadIdx.x;   // 4096 total
    const int lane = gid & 31;
    const int t    = gid >> 5;                         // 0..127
    const int j    = t >> 3, ks = t & 7;
    const int r    = lane & 15, hi = lane >> 4;
    v16h v;
    #pragma unroll
    for (int e = 0; e < 16; ++e)
        v[e] = (_Float16)W[(long)(ks * 32 + ((e < 8) ? (8 * hi + e) : (16 + 8 * hi + e - 8))) * A2_DIM + j * 16 + r];
    *(v16h*)(out + (long)gid * 16) = v;
}

// =====================================================================
// Pack X (B,N,D f32) into WMMA A-operand layout for the pooling GEMM:
//   A = X^T tile: M = d, K = n.  tile index = ((b*16 + dt)*64 + ns)
// =====================================================================
__global__ __launch_bounds__(256)
void pack_x_kernel(const float* __restrict__ X, _Float16* __restrict__ out)
{
    const long gid  = (long)blockIdx.x * 256 + threadIdx.x;  // 262144 total
    const int lane  = (int)(gid & 31);
    const int ns    = (int)((gid >> 5) & 63);
    const int dt    = (int)((gid >> 11) & 15);
    const int b     = (int)(gid >> 15);
    const int r     = lane & 15, hi = lane >> 4;
    v16h v;
    #pragma unroll
    for (int e = 0; e < 16; ++e) {
        const int vq = e >> 1, h = e & 1;
        const int kk = a_k(vq, hi) + h;
        v[e] = (_Float16)X[((long)b * N_DIM + ns * 32 + kk) * D_DIM + dt * 16 + r];
    }
    *(v16h*)(out + gid * 16) = v;
}

// =====================================================================
// Kernel 1: fused token MLP + heads (packed-f16 weights).
// =====================================================================
__global__ __launch_bounds__(128)
void mlp_head_kernel(const float* __restrict__ X,
                     const _Float16* __restrict__ pW1, const float* __restrict__ b1,
                     const _Float16* __restrict__ pW2, const float* __restrict__ b2,
                     const float* __restrict__ w_mu, const float* __restrict__ b_mu,
                     const float* __restrict__ w_sig, const float* __restrict__ b_sig,
                     float* __restrict__ mu_out, float* __restrict__ sh_out)
{
    __shared__ _Float16 h1[4][16][A2_DIM];   // 32 KB: per-wave 16x256 staging

    const int tid  = threadIdx.x;
    const int wave = tid >> 5;
    const int lane = tid & 31;
    const int r    = lane & 15;
    const int hi   = lane >> 4;
    const long m0  = ((long)blockIdx.x * 4 + wave) * 16;   // token base

    // ---- preload A tiles (16 tokens x K=256) as f16, 8 k-steps of 32 ----
    v16h a1[8];
    {
        const float* xrow = X + (m0 + r) * D_DIM;
        #pragma unroll
        for (int ks = 0; ks < 8; ++ks) {
            #pragma unroll
            for (int v = 0; v < 8; ++v) {
                const int k0 = ks * 32 + a_k(v, hi);
                const float2 xv = *(const float2*)(xrow + k0);
                a1[ks][2 * v]     = (_Float16)xv.x;
                a1[ks][2 * v + 1] = (_Float16)xv.y;
            }
        }
    }

    // ---- layer 1: 16 N-tiles, vector B loads from packed W1, gelu->LDS ----
    #pragma unroll 1
    for (int j = 0; j < 16; ++j) {
        v8f c = {};
        #pragma unroll
        for (int ks = 0; ks < 8; ++ks) {
            const v16h bt = *(const v16h*)(pW1 + ((long)((j * 8 + ks) * 32 + lane)) * 16);
            c = __builtin_amdgcn_wmma_f32_16x16x32_f16(false, a1[ks], false, bt,
                                                       (short)0, c, false, false);
        }
        const float bias = b1[j * 16 + r];
        #pragma unroll
        for (int p = 0; p < 8; ++p)
            h1[wave][p + 8 * hi][j * 16 + r] = (_Float16)gelu_exact(c[p] + bias);
    }
    __syncthreads();

    // ---- reload layer-2 A tiles from LDS: 2 x b128 per k-step ----
    v16h a2[8];
    {
        const _Float16* row = &h1[wave][r][0];
        #pragma unroll
        for (int ks = 0; ks < 8; ++ks) {
            const v8h lo  = *(const v8h*)(row + ks * 32 + 8 * hi);
            const v8h hi8 = *(const v8h*)(row + ks * 32 + 16 + 8 * hi);
            a2[ks] = cat8(lo, hi8);
        }
    }

    // ---- layer 2 + fused mu / sigma heads ----
    v8f amu = {}, asg = {};
    #pragma unroll 1
    for (int j = 0; j < 16; ++j) {
        v8f c = {};
        #pragma unroll
        for (int ks = 0; ks < 8; ++ks) {
            const v16h bt = *(const v16h*)(pW2 + ((long)((j * 8 + ks) * 32 + lane)) * 16);
            c = __builtin_amdgcn_wmma_f32_16x16x32_f16(false, a2[ks], false, bt,
                                                       (short)0, c, false, false);
        }
        const float bias = b2[j * 16 + r];
        const float wm   = w_mu[j * 16 + r];
        const float ws   = w_sig[j * 16 + r];
        #pragma unroll
        for (int p = 0; p < 8; ++p) {
            const float g = gelu_exact(c[p] + bias);
            amu[p] += g * wm;
            asg[p] += g * ws;
        }
    }

    // reduce over the 16 lanes of each half (rows differ between halves)
    #pragma unroll
    for (int off = 1; off < 16; off <<= 1) {
        #pragma unroll
        for (int p = 0; p < 8; ++p) {
            amu[p] += __shfl_xor(amu[p], off, 32);
            asg[p] += __shfl_xor(asg[p], off, 32);
        }
    }
    if (r == 0) {
        const float bmu = b_mu[0], bsg = b_sig[0];
        #pragma unroll
        for (int p = 0; p < 8; ++p) {
            const long t = m0 + p + 8 * hi;
            mu_out[t] = amu[p] + bmu;
            sh_out[t] = expf(0.5f * (asg[p] + bsg));
        }
    }
}

// =====================================================================
// Kernel 2 (fused): per-block softmax stats + pooling GEMM.
// Prob tile stored [s][k] so B-operand reads are 2 x ds_load_b128 and
// fill commits are 1 x ds_store_b128 per thread.
// =====================================================================
template<bool PACKED>
__global__ __launch_bounds__(512)
void pool_gemm_kernel(const float* __restrict__ X, const _Float16* __restrict__ pX,
                      const float* __restrict__ eps, const unsigned char* __restrict__ mask,
                      const float* __restrict__ mu, const float* __restrict__ sh,
                      float* __restrict__ Z)
{
    __shared__ _Float16 pl[SCH][KPAD];      // s_local x k_local (padded rows)
    __shared__ float red[8][64];
    __shared__ float m_sm[64];
    __shared__ float il_sm[64];

    const int b    = blockIdx.y;
    const int sc   = blockIdx.x;            // s-chunk index
    const int tid  = threadIdx.x;
    const int wave = tid >> 5;
    const int lane = tid & 31;
    const int r    = lane & 15;
    const int hi   = lane >> 4;
    const int d0   = wave * 16;             // this wave's d-tile

    const float* __restrict__ Xb   = X   + (long)b * N_DIM * D_DIM;
    const float* __restrict__ epsb = eps + (long)b * N_DIM * S_DIM;
    const float* __restrict__ mub  = mu  + (long)b * N_DIM;
    const float* __restrict__ shb  = sh  + (long)b * N_DIM;
    const unsigned char* __restrict__ mb = mask + (long)b * N_DIM;

    const int fs     = tid & 63;            // s_local (coalesced eps reads)
    const int fk0    = tid >> 6;            // 0..7 -> k-octet
    const int s_fill = sc * SCH + fs;
    const bool s_ok  = (s_fill < S_DIM);

    // ---------------- phase 0: softmax stats for 64 columns ----------------
    {
        const int nlo = fk0 * (N_DIM / 8), nhi = nlo + (N_DIM / 8);
        float mx = -3.0e38f;
        if (s_ok) {
            for (int n = nlo; n < nhi; ++n) {
                if (mb[n]) {
                    float f = fmaf(shb[n], epsb[(long)n * S_DIM + s_fill], mub[n]);
                    f = fminf(20.0f, fmaxf(-20.0f, f));
                    mx = fmaxf(mx, f);
                }
            }
        }
        red[fk0][fs] = mx;
        __syncthreads();
        if (tid < 64) {
            float m2 = red[0][tid];
            #pragma unroll
            for (int i = 1; i < 8; ++i) m2 = fmaxf(m2, red[i][tid]);
            m_sm[tid] = m2;
        }
        __syncthreads();
        const float mc = m_sm[fs];
        float sum = 0.0f;
        if (s_ok) {
            for (int n = nlo; n < nhi; ++n) {
                if (mb[n]) {
                    float f = fmaf(shb[n], epsb[(long)n * S_DIM + s_fill], mub[n]);
                    f = fminf(20.0f, fmaxf(-20.0f, f));
                    sum += expf(f - mc);
                }
            }
        }
        red[fk0][fs] = sum;
        __syncthreads();
        if (tid < 64) {
            float l = red[0][tid];
            #pragma unroll
            for (int i = 1; i < 8; ++i) l += red[i][tid];
            il_sm[tid] = 1.0f / l;
        }
        __syncthreads();
    }

    const float mcol = m_sm[fs];
    v8f acc[4] = {};

    // ---------------- main loop over 32 n-chunks of 64 ----------------
    #pragma unroll 1
    for (int ch = 0; ch < N_DIM / 64; ++ch) {
        const int nb = ch * 64;

        // ---- build probability tile exp(clip(mu+sh*eps)-m): one b128/thread ----
        {
            v8h ph;
            #pragma unroll
            for (int i = 0; i < 8; ++i) {
                const int n = nb + fk0 * 8 + i;
                float p = 0.0f;
                if (s_ok && mb[n]) {
                    float f = fmaf(shb[n], epsb[(long)n * S_DIM + s_fill], mub[n]);
                    f = fminf(20.0f, fmaxf(-20.0f, f));
                    p = expf(f - mcol);
                }
                ph[i] = (_Float16)p;
                if (ch + 1 < N_DIM / 64)   // stream-prefetch next chunk of eps
                    __builtin_prefetch(epsb + (long)(n + 64) * S_DIM + s_fill, 0, 0);
            }
            *(v8h*)&pl[fs][fk0 * 8] = ph;
        }
        __syncthreads();

        // ---- WMMA: A = X^T tile (d x k), B = prob tile (k x s) ----
        #pragma unroll
        for (int ks = 0; ks < 2; ++ks) {
            v16h at;
            if (PACKED) {
                const long tidx = (((long)b * 16 + wave) * 64 + (ch * 2 + ks)) * 32 + lane;
                at = *(const v16h*)(pX + tidx * 16);
            } else {
                #pragma unroll
                for (int v = 0; v < 8; ++v) {
                    const int n0 = nb + ks * 32 + a_k(v, hi);
                    at[2 * v]     = (_Float16)Xb[(long)n0 * D_DIM + d0 + r];
                    at[2 * v + 1] = (_Float16)Xb[(long)(n0 + 1) * D_DIM + d0 + r];
                }
            }
            #pragma unroll
            for (int ss = 0; ss < 4; ++ss) {
                // lane's 16 B halves are K = 8*hi+(0..7), 16+8*hi+(0..7) in the [s][k] layout
                const _Float16* src = &pl[ss * 16 + r][ks * 32 + 8 * hi];
                const v16h bt = cat8(*(const v8h*)src, *(const v8h*)(src + 16));
                acc[ss] = __builtin_amdgcn_wmma_f32_16x16x32_f16(false, at, false, bt,
                                                                 (short)0, acc[ss], false, false);
            }
        }
        __syncthreads();
    }

    // ---- scale by 1/l and store z[b,d,s] ----
    #pragma unroll
    for (int ss = 0; ss < 4; ++ss) {
        const int s = sc * SCH + ss * 16 + r;
        if (s < S_DIM) {
            const float il = il_sm[ss * 16 + r];
            #pragma unroll
            for (int p = 0; p < 8; ++p) {
                const int d = d0 + p + 8 * hi;
                Z[((long)b * D_DIM + d) * S_DIM + s] = acc[ss][p] * il;
            }
        }
    }
}

// =====================================================================
extern "C" void kernel_launch(void* const* d_in, const int* in_sizes, int n_in,
                              void* d_out, int out_size, void* d_ws, size_t ws_size,
                              hipStream_t stream)
{
    const float*         X     = (const float*)d_in[0];
    const unsigned char* mask  = (const unsigned char*)d_in[1];
    const float*         eps   = (const float*)d_in[2];
    const float*         W1    = (const float*)d_in[3];
    const float*         b1    = (const float*)d_in[4];
    const float*         W2    = (const float*)d_in[5];
    const float*         b2    = (const float*)d_in[6];
    const float*         w_mu  = (const float*)d_in[7];
    const float*         b_mu  = (const float*)d_in[8];
    const float*         w_sig = (const float*)d_in[9];
    const float*         b_sig = (const float*)d_in[10];
    float*               Z     = (float*)d_out;

    // ---- workspace layout ----
    char* ws = (char*)d_ws;
    float* mu_ws = (float*)ws;                                   // 16384 f32
    float* sh_ws = mu_ws + (long)B_DIM * N_DIM;                  // 16384 f32
    _Float16* pW1 = (_Float16*)(sh_ws + (long)B_DIM * N_DIM);    // 65536 f16
    _Float16* pW2 = pW1 + (long)D_DIM * A2_DIM;                  // 65536 f16
    _Float16* pX  = pW2 + (long)D_DIM * A2_DIM;                  // 4M f16 (8 MB)

    const size_t base_bytes = 2 * (size_t)B_DIM * N_DIM * 4
                            + 2 * (size_t)D_DIM * A2_DIM * 2;
    const size_t packx_bytes = (size_t)B_DIM * N_DIM * D_DIM * 2;
    const bool use_packx = (ws_size >= base_bytes + packx_bytes);

    // 0) one-time repacks into WMMA-native f16 layouts
    pack_w_kernel<<<dim3(16), dim3(256), 0, stream>>>(W1, pW1);
    pack_w_kernel<<<dim3(16), dim3(256), 0, stream>>>(W2, pW2);
    if (use_packx)
        pack_x_kernel<<<dim3(1024), dim3(256), 0, stream>>>(X, pX);

    // 1) fused MLP + heads: 16384 tokens / (4 waves * 16 tokens) = 256 blocks
    mlp_head_kernel<<<dim3(256), dim3(128), 0, stream>>>(
        X, pW1, b1, pW2, b2, w_mu, b_mu, w_sig, b_sig, mu_ws, sh_ws);

    // 2) fused softmax-stats + pooling GEMM
    if (use_packx)
        pool_gemm_kernel<true><<<dim3((S_DIM + SCH - 1) / SCH, B_DIM), dim3(512), 0, stream>>>(
            X, pX, eps, mask, mu_ws, sh_ws, Z);
    else
        pool_gemm_kernel<false><<<dim3((S_DIM + SCH - 1) / SCH, B_DIM), dim3(512), 0, stream>>>(
            X, pX, eps, mask, mu_ws, sh_ws, Z);
}
